// _Interactions_79791902425117
// MI455X (gfx1250) — matrix operand order repaired and checked by reading, not compiled, not run on hardware
//
#include <hip/hip_runtime.h>

// GCN2-style network on MI455X (gfx1250, wave32, WMMA).
// Pipeline per kernel_launch:
//   1) deg (atomic scatter) -> dinv -> edge norm           (fp32, L2 atomics)
//   2) convert weights once: W[k][n] -> f16 Bt[n][k]       (contig B-frag loads)
//   3) x0 = relu(x @ W0 + b) via v_wmma_f32_16x16x32_f16   (f32 accum)
//   4) 3x: zero agg; scatter-add norm*out[row] -> agg[col];
//          a16 = f16(0.1*agg + 0.9*x0); out += relu(a16 @ Wl)

#define C_DIM 256   // feature width (matches reference C=256)

typedef __attribute__((ext_vector_type(16))) _Float16 v16h;
typedef __attribute__((ext_vector_type(8)))  _Float16 v8h;
typedef __attribute__((ext_vector_type(8)))  float    v8f;

// ---------------- small elementwise / graph kernels ----------------

__global__ void k_zero(float* __restrict__ p, size_t n) {
  size_t i = (size_t)blockIdx.x * blockDim.x + threadIdx.x;
  if (i < n) p[i] = 0.0f;
}

__global__ void k_deg(const int* __restrict__ ei, const float* __restrict__ w,
                      float* __restrict__ deg, int E) {
  int e = blockIdx.x * blockDim.x + threadIdx.x;
  if (e < E) atomicAdd(&deg[ei[E + e]], w[e]);   // segment-sum by col
}

__global__ void k_dinv(float* __restrict__ deg, int N) {
  int i = blockIdx.x * blockDim.x + threadIdx.x;
  if (i < N) { float d = deg[i]; deg[i] = (d > 0.0f) ? rsqrtf(d) : 0.0f; }
}

__global__ void k_norm(const int* __restrict__ ei, const float* __restrict__ w,
                       const float* __restrict__ dinv, float* __restrict__ nrm, int E) {
  int e = blockIdx.x * blockDim.x + threadIdx.x;
  if (e < E) nrm[e] = dinv[ei[e]] * w[e] * dinv[ei[E + e]];
}

// Transpose + convert all (1+L) weight matrices: w16[mat][n][k] = (f16)W[mat][k][n]
__global__ void k_cvt_w(const float* __restrict__ lin0_w, const float* __restrict__ conv_w,
                        _Float16* __restrict__ w16, int nmats) {
  int tid = blockIdx.x * blockDim.x + threadIdx.x;
  int total = nmats * C_DIM * C_DIM;
  if (tid >= total) return;
  int mat = tid / (C_DIM * C_DIM);
  int rem = tid - mat * (C_DIM * C_DIM);
  int k = rem / C_DIM, n = rem - k * C_DIM;
  float v = (mat == 0) ? lin0_w[rem]
                       : conv_w[(size_t)(mat - 1) * C_DIM * C_DIM + rem];
  w16[(size_t)mat * C_DIM * C_DIM + (size_t)n * C_DIM + k] = (_Float16)v;
}

__global__ void k_cvt(const float* __restrict__ x, _Float16* __restrict__ a16, size_t n) {
  size_t i = (size_t)blockIdx.x * blockDim.x + threadIdx.x;
  if (i < n) a16[i] = (_Float16)x[i];
}

// sup = (1-alpha)*agg + alpha*x0, converted to f16 GEMM input. alpha = 0.9
__global__ void k_mix(const float* __restrict__ agg, const float* __restrict__ x0,
                      _Float16* __restrict__ a16, size_t n) {
  size_t i = (size_t)blockIdx.x * blockDim.x + threadIdx.x;
  if (i < n) a16[i] = (_Float16)(0.1f * agg[i] + 0.9f * x0[i]);
}

// agg[col] += norm_e * out[row] ; 64 consecutive threads cover one edge's 256 ch.
__global__ void k_scatter(const int* __restrict__ ei, const float* __restrict__ nrm,
                          const float* __restrict__ src, float* __restrict__ agg, int E) {
  long tid = (long)blockIdx.x * blockDim.x + threadIdx.x;
  long e = tid >> 6;                 // / (C_DIM/4)
  if (e >= (long)E) return;
  int c = (int)(tid & 63) * 4;
  int r   = ei[e];
  int col = ei[(long)E + e];
  float nv = nrm[e];
  float4 v = *(const float4*)(src + (size_t)r * C_DIM + c);
  float* dp = agg + (size_t)col * C_DIM + c;
  atomicAdd(dp + 0, nv * v.x);
  atomicAdd(dp + 1, nv * v.y);
  atomicAdd(dp + 2, nv * v.z);
  atomicAdd(dp + 3, nv * v.w);
}

// ---------------- WMMA GEMM ----------------
// D[M x 256] = A[M x 256] (f16) @ W (via Bt[n][k] f16), f32 accumulate.
// Block: 256 threads = 8 waves; block tile 16 rows x 128 cols, wave tile 16x16.
// mode 0: v = relu(acc + bias[n]); x0[.]=v; out[.]=v        (lin0)
// mode 1: out[.] += relu(acc)                               (conv layer)
__global__ __launch_bounds__(256) void k_gemm(
    const _Float16* __restrict__ A,    // [M][256] row-major f16
    const _Float16* __restrict__ Bt,   // [256][256] f16, Bt[n][k] = W[k][n]
    const float*    __restrict__ bias, // C floats (mode 0) or unused
    float*          __restrict__ out,  // [M][256] f32
    float*          __restrict__ x0,   // [M][256] f32 (mode 0)
    int M, int mode) {
  const int lane = threadIdx.x & 31;
  const int wave = threadIdx.x >> 5;
  const int row0 = blockIdx.x * 16;
  const int col0 = blockIdx.y * 128 + wave * 16;

  const int m  = lane & 15;   // row within A tile / col within B tile
  const int hs = lane >> 4;   // K half select (contiguous k = 16*hs + s, same order for A and B)

  int arow = row0 + m; if (arow > M - 1) arow = M - 1;   // keep EXEC all-ones
  const _Float16* ap = A  + (size_t)arow * C_DIM;
  const _Float16* bp = Bt + (size_t)(col0 + m) * C_DIM + hs * 16;

  v8f acc = {};
#pragma unroll
  for (int kk = 0; kk < C_DIM; kk += 32) {
    // A frag: lanes 0-15 K=kk+0..15 ; lanes 16-31 K=kk+16..31 (contiguous, matches B)
    v8h alo = *(const v8h*)(ap + kk + hs * 16);
    v8h ahi = *(const v8h*)(ap + kk + hs * 16 + 8);
    // B frag: lane's column, K striped through the 16 halves (contiguous in Bt)
    v8h b0  = *(const v8h*)(bp + kk);
    v8h b1  = *(const v8h*)(bp + kk + 8);
    v16h a, b;
#pragma unroll
    for (int i = 0; i < 8; ++i) {
      a[i] = alo[i]; a[i + 8] = ahi[i];
      b[i] = b0[i];  b[i + 8] = b1[i];
    }
    acc = __builtin_amdgcn_wmma_f32_16x16x32_f16(
        /*neg_a=*/false, a, /*neg_b=*/false, b,
        /*c_mod=*/(short)0, acc, /*reuse_a=*/false, /*reuse_b=*/false);
  }

  // C/D layout: VGPR r -> M=r (lanes 0-15) / M=r+8 (lanes 16-31), N=lane%16
  const int ncol  = col0 + m;
  const int mbase = row0 + (hs ? 8 : 0);
#pragma unroll
  for (int r = 0; r < 8; ++r) {
    int mrow = mbase + r;
    if (mrow < M) {
      size_t idx = (size_t)mrow * C_DIM + ncol;
      float v = acc[r];
      if (mode == 0) {
        v += bias[ncol];
        v = v > 0.0f ? v : 0.0f;
        x0[idx]  = v;
        out[idx] = v;
      } else {
        v = v > 0.0f ? v : 0.0f;
        out[idx] = out[idx] + v;
      }
    }
  }
}

// ---------------- host-side orchestration ----------------

static inline int cdiv(long a, long b) { return (int)((a + b - 1) / b); }

extern "C" void kernel_launch(void* const* d_in, const int* in_sizes, int n_in,
                              void* d_out, int out_size, void* d_ws, size_t ws_size,
                              hipStream_t stream) {
  const float* x       = (const float*)d_in[0];
  const int*   ei      = (const int*)  d_in[1];
  const float* ew      = (const float*)d_in[2];
  /* d_in[3] = edge_attr : unused by forward */
  const float* lin0_w  = (const float*)d_in[4];
  const float* lin0_b  = (const float*)d_in[5];
  const float* conv_w  = (const float*)d_in[6];

  const int N = in_sizes[0] / C_DIM;
  const int E = in_sizes[2];
  const int L = in_sizes[6] / (C_DIM * C_DIM);
  const size_t NC = (size_t)N * C_DIM;

  // Carve workspace (256B aligned slices)
  char* ws = (char*)d_ws;
  size_t off = 0;
  auto carve = [&](size_t bytes) -> char* {
    char* p = ws + off;
    off = (off + bytes + 255) & ~(size_t)255;
    return p;
  };
  float*    deg = (float*)   carve((size_t)N * 4);        // -> dinv in place
  float*    nrm = (float*)   carve((size_t)E * 4);
  float*    x0  = (float*)   carve(NC * 4);
  float*    agg = (float*)   carve(NC * 4);
  _Float16* a16 = (_Float16*)carve(NC * 2);
  _Float16* w16 = (_Float16*)carve((size_t)(1 + L) * C_DIM * C_DIM * 2);
  (void)ws_size;

  float* out = (float*)d_out;   // running `out` across layers
  const int TB = 256;

  // 1) normalization coefficients
  k_zero<<<cdiv(N, TB), TB, 0, stream>>>(deg, (size_t)N);
  k_deg <<<cdiv(E, TB), TB, 0, stream>>>(ei, ew, deg, E);
  k_dinv<<<cdiv(N, TB), TB, 0, stream>>>(deg, N);
  k_norm<<<cdiv(E, TB), TB, 0, stream>>>(ei, ew, deg, nrm, E);

  // 2) weights -> transposed f16 (once)
  k_cvt_w<<<cdiv((long)(1 + L) * C_DIM * C_DIM, TB), TB, 0, stream>>>(
      lin0_w, conv_w, w16, 1 + L);

  // 3) x0 = relu(x @ W0 + b); out = x0
  k_cvt<<<cdiv((long)NC, TB), TB, 0, stream>>>(x, a16, NC);
  {
    dim3 grid(cdiv(N, 16), C_DIM / 128);
    k_gemm<<<grid, TB, 0, stream>>>(a16, w16, lin0_b, out, x0, N, /*mode=*/0);
  }

  // 4) layers
  for (int li = 0; li < L; ++li) {
    k_zero<<<cdiv((long)NC, TB), TB, 0, stream>>>(agg, NC);
    k_scatter<<<cdiv((long)E * (C_DIM / 4), TB), TB, 0, stream>>>(ei, nrm, out, agg, E);
    k_mix<<<cdiv((long)NC, TB), TB, 0, stream>>>(agg, x0, a16, NC);
    dim3 grid(cdiv(N, 16), C_DIM / 128);
    k_gemm<<<grid, TB, 0, stream>>>(a16, w16 + (size_t)(1 + li) * C_DIM * C_DIM,
                                    lin0_b /*unused*/, out, x0 /*unused*/, N, /*mode=*/1);
  }
}
